// CrossFusion_82832739270988
// MI455X (gfx1250) — compile-verified, matrix operand order checked by reading, not run on hardware
//
#include <hip/hip_runtime.h>
#include <hip/hip_bf16.h>
#include <stdint.h>

// ---------------------------------------------------------------------------
// CDNA5 (gfx1250) bf16 WMMA implementation of the CLIP<-LLM cross-attention
// fusion block.  All matmuls are X @ W^T form -> both A and B fragments are
// K-contiguous b128 loads matching the wave32 WMMA VGPR layouts.
// GEMM: 128x128 block tile, 8 waves (4x2), wave tile 32x64 -> 8 WMMAs per
// wave per K-step(32) between barriers, with register double-buffered
// global->LDS staging so global loads overlap WMMA issue.
// ---------------------------------------------------------------------------

typedef __bf16 bf16;
typedef bf16 v16bf __attribute__((ext_vector_type(16)));
typedef bf16 v8bf  __attribute__((ext_vector_type(8)));
typedef float v8f  __attribute__((ext_vector_type(8)));

#define DEV __device__ __forceinline__

static constexpr int B_   = 64;
static constexpr int LC_  = 77;
static constexpr int LL_  = 256;
static constexpr int LDIM = 2048;
static constexpr int DIM_ = 768;
static constexpr int HDIM = 96;
static constexpr float SCALE_ = 0.10206207261596575f; // 96^-0.5

DEV bf16 f2bf(float f) {
  union { float f; uint32_t u; } v; v.f = f;
  uint32_t r = v.u + 0x7FFFu + ((v.u >> 16) & 1u);   // RNE
  union { uint16_t u; bf16 b; } o; o.u = (uint16_t)(r >> 16);
  return o.b;
}

// A-matrix 16x32 bf16 fragment (also B 32x16 via symmetric [N][K] layout):
// lane L: row = row_off + (L&15) (clamped), half = L>>4
// VGPR 0..3 : k = half*8 + 0..7   (16 contiguous bytes)
// VGPR 4..7 : k = 16 + half*8 + 0..7
DEV v16bf ldfrag(const bf16* base, int row_off, int stride, int lane, int row_max) {
  int r = row_off + (lane & 15);
  r = (r < row_max) ? r : row_max;
  const bf16* p = base + (size_t)r * stride + (lane >> 4) * 8;
  v8bf lo = *(const v8bf*)(p);
  v8bf hi = *(const v8bf*)(p + 16);
  return __builtin_shufflevector(lo, hi, 0,1,2,3,4,5,6,7,8,9,10,11,12,13,14,15);
}

DEV v8f wmma_bf16(v16bf a, v16bf b, v8f c) {
  return __builtin_amdgcn_wmma_f32_16x16x32_bf16(false, a, false, b, (short)0, c,
                                                 false, false);
}

// ---------------------------------------------------------------------------
// Generic GEMM: out[m,n] = sum_k A[m,k] * W[n,k] (+ bias / residual / act)
// EPI: 0 f32 store | 1 bf16 store | 2 f32 store + residual
//      3 bf16 store + QuickGELU   | 4 f32 store + residual, rows remapped
//                                   into the concat output (b*333 + 77 + l)
// ---------------------------------------------------------------------------
template <int EPI>
__global__ __launch_bounds__(256) void gemm_xwt(
    const bf16* __restrict__ A, const bf16* __restrict__ W,
    const float* __restrict__ bias, const float* __restrict__ resid,
    float* __restrict__ outF, bf16* __restrict__ outB,
    int M, int N, int K) {
  __shared__ __align__(16) bf16 As[128 * 32];
  __shared__ __align__(16) bf16 Bs[128 * 32];

  const int tid  = threadIdx.x;
  const int lane = tid & 31, wv = tid >> 5;
  const int wm = wv >> 1, wn = wv & 1;          // 4 x 2 wave grid
  const int bm = blockIdx.y * 128, bn = blockIdx.x * 128;

  // cooperative staging: 2 threads per tile row, 16 bf16 (2 x b128) each
  const int lr = tid >> 1, lc = (tid & 1) * 16;
  const int gr = (bm + lr < M) ? (bm + lr) : (M - 1);   // clamp ragged M
  const bf16* ap = A + (size_t)gr * K + lc;
  const bf16* wp = W + (size_t)(bn + lr) * K + lc;

  v8f acc[2][4] = {};
  v8bf sa0, sa1, sb0, sb1;                      // double-buffer staging regs

  sa0 = *(const v8bf*)(ap);     sa1 = *(const v8bf*)(ap + 8);
  sb0 = *(const v8bf*)(wp);     sb1 = *(const v8bf*)(wp + 8);

  for (int kt = 0; kt < K; kt += 32) {
    *(v8bf*)&As[lr * 32 + lc]     = sa0;
    *(v8bf*)&As[lr * 32 + lc + 8] = sa1;
    *(v8bf*)&Bs[lr * 32 + lc]     = sb0;
    *(v8bf*)&Bs[lr * 32 + lc + 8] = sb1;
    __syncthreads();

    if (kt + 32 < K) {          // issue next-tile global loads under the WMMAs
      sa0 = *(const v8bf*)(ap + kt + 32);
      sa1 = *(const v8bf*)(ap + kt + 40);
      sb0 = *(const v8bf*)(wp + kt + 32);
      sb1 = *(const v8bf*)(wp + kt + 40);
      if (kt + 64 < K) {        // gfx1250 global_prefetch_b8
        __builtin_prefetch(ap + kt + 64, 0, 0);
        __builtin_prefetch(wp + kt + 64, 0, 0);
      }
    }

    v16bf fa0 = ldfrag(As, wm * 32,      32, lane, 1 << 30);
    v16bf fa1 = ldfrag(As, wm * 32 + 16, 32, lane, 1 << 30);
#pragma unroll
    for (int j = 0; j < 4; ++j) {
      v16bf fb = ldfrag(Bs, wn * 64 + j * 16, 32, lane, 1 << 30);
      acc[0][j] = wmma_bf16(fa0, fb, acc[0][j]);
      acc[1][j] = wmma_bf16(fa1, fb, acc[1][j]);
    }
    __syncthreads();
  }

  // C/D layout: lane L -> col = (L&15); VGPR v -> row = v + (L>>4)*8
  const int nn = lane & 15, hh = lane >> 4;
  for (int i = 0; i < 2; ++i)
    for (int j = 0; j < 4; ++j) {
      const int col = bn + wn * 64 + j * 16 + nn;
      const float bv = bias[col];
      for (int v = 0; v < 8; ++v) {
        const int row = bm + wm * 32 + i * 16 + hh * 8 + v;
        if (row >= M) continue;
        float x = acc[i][j][v] + bv;
        if (EPI == 0) {
          outF[(size_t)row * N + col] = x;
        } else if (EPI == 1) {
          outB[(size_t)row * N + col] = f2bf(x);
        } else if (EPI == 2) {
          outF[(size_t)row * N + col] = x + resid[(size_t)row * N + col];
        } else if (EPI == 3) {
          float g = x / (1.f + __expf(-1.702f * x));   // QuickGELU
          outB[(size_t)row * N + col] = f2bf(g);
        } else {
          float r2 = resid[(size_t)row * N + col];
          int orow = (row >> 8) * (LC_ + LL_) + LC_ + (row & 255);
          outF[(size_t)orow * N + col] = x + r2;
        }
      }
    }
}

// ---------------------------------------------------------------------------
// Cross-attention, one block per (b, h, 64-query chunk).
// Scores: Q/K fragments straight from global (K-token clamped to 76),
// masked softmax over 77 CLIP tokens, probs bf16 (K padded 77->96 w/ zeros),
// V^T staged in LDS, second WMMA pass for P @ V.
// ---------------------------------------------------------------------------
__global__ __launch_bounds__(256) void cross_attn(
    const bf16* __restrict__ Q, const bf16* __restrict__ Kb,
    const bf16* __restrict__ Vb, bf16* __restrict__ O) {
  __shared__ __align__(16) float Ssm[64 * 80];
  __shared__ __align__(16) bf16  Pbuf[64 * 96];
  __shared__ __align__(16) bf16  Vt[96 * 96];

  const int tid = threadIdx.x, lane = tid & 31, wv = tid >> 5;
  const int mc = blockIdx.x, h = blockIdx.y, b = blockIdx.z;
  const int qr0 = b * LL_ + mc * 64;
  const size_t qoff = (size_t)qr0 * DIM_ + h * HDIM;
  const size_t koff = (size_t)b * LC_ * DIM_ + h * HDIM;

  // V^T tile [d][j], zero-padded j in [77,96)
  for (int idx = tid; idx < 96 * 96; idx += 256) {
    int d = idx / 96, j = idx % 96;
    Vt[idx] = (j < LC_) ? Vb[koff + (size_t)j * DIM_ + d] : f2bf(0.f);
  }

  const int nn = lane & 15, hh = lane >> 4;

  // scores: 64x80 = 4x5 tiles over K=96 (3 WMMA steps)
  for (int t = wv; t < 20; t += 8) {
    int tm = t / 5, tn = t % 5;
    v8f acc = {};
    for (int ks = 0; ks < HDIM; ks += 32) {
      v16bf fq = ldfrag(Q + qoff + ks, tm * 16, DIM_, lane, 1 << 30);
      v16bf fk = ldfrag(Kb + koff + ks, tn * 16, DIM_, lane, LC_ - 1);
      acc = wmma_bf16(fq, fk, acc);
    }
    for (int v = 0; v < 8; ++v) {
      int r = tm * 16 + hh * 8 + v, c = tn * 16 + nn;
      Ssm[r * 80 + c] = acc[v] * SCALE_;
    }
  }
  __syncthreads();

  // masked softmax, one row per thread
  if (tid < 64) {
    float mx = -3.4e38f;
    for (int j = 0; j < LC_; ++j) mx = fmaxf(mx, Ssm[tid * 80 + j]);
    float sum = 0.f;
    for (int j = 0; j < LC_; ++j) {
      float e = __expf(Ssm[tid * 80 + j] - mx);
      Ssm[tid * 80 + j] = e;
      sum += e;
    }
    float inv = 1.f / sum;
    for (int j = 0; j < 96; ++j)
      Pbuf[tid * 96 + j] = (j < LC_) ? f2bf(Ssm[tid * 80 + j] * inv) : f2bf(0.f);
  }
  __syncthreads();

  // out = P(64x96) @ V : 4x6 tiles over K=96
  for (int t = wv; t < 24; t += 8) {
    int tm = t / 6, tn = t % 6;
    v8f acc = {};
    for (int ks = 0; ks < 96; ks += 32) {
      v16bf fp = ldfrag(Pbuf + ks, tm * 16, 96, lane, 1 << 30);
      v16bf fv = ldfrag(Vt + ks, tn * 16, 96, lane, 1 << 30);
      acc = wmma_bf16(fp, fv, acc);
    }
    for (int v = 0; v < 8; ++v) {
      int r = qr0 + tm * 16 + hh * 8 + v;
      int c = h * HDIM + tn * 16 + nn;
      O[(size_t)r * DIM_ + c] = f2bf(acc[v]);
    }
  }
}

// ---------------------------------------------------------------------------
// Support kernels
// ---------------------------------------------------------------------------
__global__ __launch_bounds__(256) void layernorm_to_bf16(
    const float* __restrict__ X, const float* __restrict__ g,
    const float* __restrict__ bt, bf16* __restrict__ Y) {
  __shared__ float rs[256], rs2[256];
  const int row = blockIdx.x, tid = threadIdx.x;
  const float* x = X + (size_t)row * DIM_;
  float s = 0.f, s2 = 0.f;
  for (int c = tid; c < DIM_; c += 256) { float v = x[c]; s += v; s2 += v * v; }
  rs[tid] = s; rs2[tid] = s2;
  __syncthreads();
  for (int o = 128; o > 0; o >>= 1) {
    if (tid < o) { rs[tid] += rs[tid + o]; rs2[tid] += rs2[tid + o]; }
    __syncthreads();
  }
  const float mean = rs[0] * (1.f / DIM_);
  const float var  = rs2[0] * (1.f / DIM_) - mean * mean;
  const float inv  = rsqrtf(var + 1e-5f);
  bf16* y = Y + (size_t)row * DIM_;
  for (int c = tid; c < DIM_; c += 256)
    y[c] = f2bf((x[c] - mean) * inv * g[c] + bt[c]);
}

__global__ void cvt_f32_bf16(const float* __restrict__ s, bf16* __restrict__ d,
                             size_t n) {
  size_t i = (size_t)blockIdx.x * blockDim.x + threadIdx.x;
  if (i < n) d[i] = f2bf(s[i]);
}

__global__ void copy_clip_out(const float* __restrict__ clip,
                              float* __restrict__ out) {
  size_t i = (size_t)blockIdx.x * blockDim.x + threadIdx.x;
  const size_t n = (size_t)B_ * LC_ * DIM_;
  if (i < n) {
    size_t b = i / ((size_t)LC_ * DIM_), r = i % ((size_t)LC_ * DIM_);
    out[b * (size_t)(LC_ + LL_) * DIM_ + r] = clip[i];
  }
}

// ---------------------------------------------------------------------------
extern "C" void kernel_launch(void* const* d_in, const int* in_sizes, int n_in,
                              void* d_out, int out_size, void* d_ws,
                              size_t ws_size, hipStream_t stream) {
  const float* clip_embed = (const float*)d_in[0];
  const float* llm_embed  = (const float*)d_in[1];
  const float* llm_proj_w = (const float*)d_in[2];
  const float* llm_proj_b = (const float*)d_in[3];
  const float* q_w = (const float*)d_in[4];  const float* q_b = (const float*)d_in[5];
  const float* k_w = (const float*)d_in[6];  const float* k_b = (const float*)d_in[7];
  const float* v_w = (const float*)d_in[8];  const float* v_b = (const float*)d_in[9];
  const float* o_w = (const float*)d_in[10]; const float* o_b = (const float*)d_in[11];
  const float* qn_g = (const float*)d_in[12]; const float* qn_b = (const float*)d_in[13];
  const float* kvn_g = (const float*)d_in[14]; const float* kvn_b = (const float*)d_in[15];
  const float* n_g = (const float*)d_in[16];  const float* n_b = (const float*)d_in[17];
  const float* f1_w = (const float*)d_in[18]; const float* f1_b = (const float*)d_in[19];
  const float* f2_w = (const float*)d_in[20]; const float* f2_b = (const float*)d_in[21];
  float* out = (float*)d_out;

  const int ML = B_ * LL_;   // 16384 llm rows
  const int MC = B_ * LC_;   // 4928 clip rows

  // ---- workspace carving (256B aligned) ----
  char* wsb = (char*)d_ws;
  size_t off = 0;
  auto carve = [&](size_t bytes) -> void* {
    off = (off + 255) & ~(size_t)255;
    void* p = wsb + off;
    off += bytes;
    return p;
  };
  bf16* wLLM = (bf16*)carve((size_t)DIM_ * LDIM * 2);
  bf16* wQ   = (bf16*)carve((size_t)DIM_ * DIM_ * 2);
  bf16* wK   = (bf16*)carve((size_t)DIM_ * DIM_ * 2);
  bf16* wV   = (bf16*)carve((size_t)DIM_ * DIM_ * 2);
  bf16* wO   = (bf16*)carve((size_t)DIM_ * DIM_ * 2);
  bf16* wF1  = (bf16*)carve((size_t)4 * DIM_ * DIM_ * 2);
  bf16* wF2  = (bf16*)carve((size_t)DIM_ * 4 * DIM_ * 2);
  bf16* clipn = (bf16*)carve((size_t)MC * DIM_ * 2);
  bf16* llm_e = (bf16*)carve((size_t)ML * LDIM * 2);   // dead after GEMM1
  bf16* qb    = (bf16*)carve((size_t)ML * DIM_ * 2);   // dead after attention
  bf16* kb    = (bf16*)carve((size_t)MC * DIM_ * 2);   // dead after attention
  bf16* vb    = (bf16*)carve((size_t)MC * DIM_ * 2);   // dead after attention
  bf16* llmn  = (bf16*)carve((size_t)ML * DIM_ * 2);   // dead after q GEMM
  float* resid1 = (float*)carve((size_t)ML * DIM_ * 4);
  float* resid2 = (float*)carve((size_t)ML * DIM_ * 4);
  bf16* hb      = (bf16*)carve((size_t)ML * DIM_ * 2);
  bf16* attnb = llmn;   // alias: llmn dead before attention writes
  bf16* fb    = llm_e;  // alias: FFN hidden (96MB) over llm_e+q+k+v (107MB)

  auto cvt = [&](const float* s, bf16* d, size_t n) {
    cvt_f32_bf16<<<(unsigned)((n + 255) / 256), 256, 0, stream>>>(s, d, n);
  };

  // 1) precision conversion of weights + llm embeddings
  cvt(llm_proj_w, wLLM, (size_t)DIM_ * LDIM);
  cvt(q_w, wQ, (size_t)DIM_ * DIM_);
  cvt(k_w, wK, (size_t)DIM_ * DIM_);
  cvt(v_w, wV, (size_t)DIM_ * DIM_);
  cvt(o_w, wO, (size_t)DIM_ * DIM_);
  cvt(f1_w, wF1, (size_t)4 * DIM_ * DIM_);
  cvt(f2_w, wF2, (size_t)DIM_ * 4 * DIM_);
  cvt(llm_embed, llm_e, (size_t)ML * LDIM);

  // 2) clip passthrough + layernorm(clip)
  {
    size_t n = (size_t)B_ * LC_ * DIM_;
    copy_clip_out<<<(unsigned)((n + 255) / 256), 256, 0, stream>>>(clip_embed, out);
  }
  layernorm_to_bf16<<<MC, 256, 0, stream>>>(clip_embed, qn_g, qn_b, clipn);

  // 3) llm = llm_embed @ llm_proj_w^T + b   (fp32 residual)
  gemm_xwt<0><<<dim3(DIM_ / 128, ML / 128), 256, 0, stream>>>(
      llm_e, wLLM, llm_proj_b, nullptr, resid1, nullptr, ML, DIM_, LDIM);
  layernorm_to_bf16<<<ML, 256, 0, stream>>>(resid1, kvn_g, kvn_b, llmn);

  // 4) q / k / v projections
  gemm_xwt<1><<<dim3(DIM_ / 128, ML / 128), 256, 0, stream>>>(
      llmn, wQ, q_b, nullptr, nullptr, qb, ML, DIM_, DIM_);
  gemm_xwt<1><<<dim3(DIM_ / 128, (MC + 127) / 128), 256, 0, stream>>>(
      clipn, wK, k_b, nullptr, nullptr, kb, MC, DIM_, DIM_);
  gemm_xwt<1><<<dim3(DIM_ / 128, (MC + 127) / 128), 256, 0, stream>>>(
      clipn, wV, v_b, nullptr, nullptr, vb, MC, DIM_, DIM_);

  // 5) cross attention
  cross_attn<<<dim3(4, 8, B_), 256, 0, stream>>>(qb, kb, vb, attnb);

  // 6) output projection + residual, layernorm
  gemm_xwt<2><<<dim3(DIM_ / 128, ML / 128), 256, 0, stream>>>(
      attnb, wO, o_b, resid1, resid2, nullptr, ML, DIM_, DIM_);
  layernorm_to_bf16<<<ML, 256, 0, stream>>>(resid2, n_g, n_b, hb);

  // 7) FFN: f1 + QuickGELU, then f2 + residual -> mapped straight into d_out
  gemm_xwt<3><<<dim3(4 * DIM_ / 128, ML / 128), 256, 0, stream>>>(
      hb, wF1, f1_b, nullptr, nullptr, fb, ML, 4 * DIM_, DIM_);
  gemm_xwt<4><<<dim3(DIM_ / 128, ML / 128), 256, 0, stream>>>(
      fb, wF2, f2_b, resid2, out, nullptr, ML, DIM_, 4 * DIM_);
}
